// SwitchTransformerEncoderLayer_33474975105482
// MI455X (gfx1250) — compile-verified, matrix operand order checked by reading, not run on hardware
//
#include <hip/hip_runtime.h>
#include <hip/hip_bf16.h>
#include <math.h>

#define DMODEL 512
#define NHEAD  8
#define DHEAD  64
#define HID    2048
#define NEXP   8
#define BATCH  4
#define SEQ    1024
#define NTOK   (BATCH*SEQ)   /* 4096 */
#define EPS    1e-5f

typedef __bf16 bf16;
typedef __attribute__((ext_vector_type(16))) __bf16 v16bf;
typedef __attribute__((ext_vector_type(8)))  __bf16 v8bf;
typedef __attribute__((ext_vector_type(8)))  float  v8f;
typedef __attribute__((ext_vector_type(4)))  float  v4f;

#define WMMA_BF16(A,B,C) \
  __builtin_amdgcn_wmma_f32_16x16x32_bf16(false,(A),false,(B),(short)0,(C),false,false)

// ---------------- fragment loaders (wave32 layouts, 16x16x32 bf16) ----------
// A fragment from a row-major matrix: lane holds row m = lane&15,
// K offsets kb..kb+7 and kb+16..kb+23, kb = (lane>=16)*8.
// Also used for B of Q@K^T (B = K^T of row-major K) and for LDS-transposed
// B panels ([n][k] layout, ld = 32) -> two contiguous 16B reads per lane.
__device__ inline v16bf frag_row(const bf16* p, int ld) {
  int lane = threadIdx.x & 31;
  const bf16* rp = p + (size_t)(lane & 15) * ld + ((lane >> 4) << 3);
  v16bf f;
#pragma unroll
  for (int i = 0; i < 8; ++i) f[i] = rp[i];
#pragma unroll
  for (int i = 0; i < 8; ++i) f[8 + i] = rp[16 + i];
  return f;
}

// ---------------- fp32 -> bf16 cast (8 elems/thread, vectorized) ------------
__global__ void k_cvt8(const float* __restrict__ in, bf16* __restrict__ out, int n8) {
  int i = blockIdx.x * blockDim.x + threadIdx.x;
  if (i >= n8) return;
  v4f a = *(const v4f*)(in + (size_t)i * 8);
  v4f b = *(const v4f*)(in + (size_t)i * 8 + 4);
  v8bf o;
#pragma unroll
  for (int j = 0; j < 4; ++j) { o[j] = (bf16)a[j]; o[4 + j] = (bf16)b[j]; }
  *(v8bf*)(out + (size_t)i * 8) = o;
}

// ============================================================================
// GEMM: C[M,N] = A[M,K] @ B[K,N] + bias.  Block = 256 threads (8 waves),
// 128x64 output tile.  B panel (32x64) staged to LDS *transposed* ([n][k],
// ld=32, one 16B chunk per thread) with double buffering; the panel is reused
// by all 8 waves (each owns 16 rows, 4 WMMAs per K-step).
// ============================================================================
template<bool WF, bool WB>
__global__ void k_gemm_t(const bf16* __restrict__ A, const bf16* __restrict__ B,
                         const float* __restrict__ bias,
                         float* __restrict__ Cf, bf16* __restrict__ Cb,
                         int N, int K) {
  __shared__ bf16 Bs[2][64 * 32];          // [buf][n][k], 8 KB
  int nbase = blockIdx.x * 64;
  int mbase = blockIdx.y * 128;
  int tid  = threadIdx.x;
  int wave = tid >> 5;
  int lane = tid & 31;

  int ks = tid >> 3;                       // staging row 0..31
  int cs = (tid & 7) << 3;                 // staging col 0,8,..,56

  const bf16* Arp = A + (size_t)(mbase + wave * 16 + (lane & 15)) * K
                      + ((lane >> 4) << 3);
  v8f acc0 = {}, acc1 = {}, acc2 = {}, acc3 = {};

  v8bf ch = *(const v8bf*)(B + (size_t)ks * N + nbase + cs);

  int buf = 0;
  for (int k0 = 0; k0 < K; k0 += 32) {
#pragma unroll
    for (int j = 0; j < 8; ++j) Bs[buf][(cs + j) * 32 + ks] = ch[j];
    __syncthreads();

    if (k0 + 32 < K)      // issue next panel's global load before compute
      ch = *(const v8bf*)(B + (size_t)(k0 + 32 + ks) * N + nbase + cs);

    v16bf af;
#pragma unroll
    for (int i = 0; i < 8; ++i) af[i] = Arp[k0 + i];
#pragma unroll
    for (int i = 0; i < 8; ++i) af[8 + i] = Arp[k0 + 16 + i];

    acc0 = WMMA_BF16(af, frag_row(&Bs[buf][ 0 * 32], 32), acc0);
    acc1 = WMMA_BF16(af, frag_row(&Bs[buf][16 * 32], 32), acc1);
    acc2 = WMMA_BF16(af, frag_row(&Bs[buf][32 * 32], 32), acc2);
    acc3 = WMMA_BF16(af, frag_row(&Bs[buf][48 * 32], 32), acc3);
    buf ^= 1;
  }

  int n = lane & 15;
  int mrow = mbase + wave * 16 + ((lane >> 4) << 3);
  float b0 = bias[nbase + n];
  float b1 = bias[nbase + 16 + n];
  float b2 = bias[nbase + 32 + n];
  float b3 = bias[nbase + 48 + n];
#pragma unroll
  for (int r = 0; r < 8; ++r) {
    size_t o = (size_t)(mrow + r) * N + nbase + n;
    float v0 = acc0[r] + b0, v1 = acc1[r] + b1;
    float v2 = acc2[r] + b2, v3 = acc3[r] + b3;
    if (WF) { Cf[o] = v0; Cf[o + 16] = v1; Cf[o + 32] = v2; Cf[o + 48] = v3; }
    if (WB) { Cb[o] = (bf16)v0; Cb[o + 16] = (bf16)v1;
              Cb[o + 32] = (bf16)v2; Cb[o + 48] = (bf16)v3; }
  }
}

// ============================================================================
// Grouped MoE GEMM: per expert e, C = act(A_rows @ W[e] + bias[e]).
// Same 8-wave 128x64 double-buffered tiling; A rows gathered via perm (GATHER)
// or contiguous in the permuted buffer; C rows scattered via perm (SCATTER).
// ============================================================================
template<bool WF, bool WB, bool GELU, bool GATHER, bool SCATTER>
__global__ void k_moe_gemm_t(const bf16* __restrict__ A, int lda,
                             const bf16* __restrict__ W, int K, int N,
                             const float* __restrict__ bias,
                             float* __restrict__ Cf, bf16* __restrict__ Cb, int ldc,
                             const int* __restrict__ perm,
                             const int* __restrict__ offs,
                             const int* __restrict__ cnt) {
  __shared__ bf16 Bs[2][64 * 32];
  int e = blockIdx.z;
  int c0 = offs[e];
  int count = cnt[e];
  int mbase = blockIdx.y * 128;
  if (mbase >= count) return;                 // block-uniform
  int nbase = blockIdx.x * 64;
  int tid  = threadIdx.x;
  int wave = tid >> 5;
  int lane = tid & 31;

  int ks = tid >> 3;
  int cs = (tid & 7) << 3;

  int rl = mbase + wave * 16 + (lane & 15);
  if (rl >= count) rl = count - 1;            // clamp pad rows (stores guarded)
  int arow = GATHER ? perm[c0 + rl] : (c0 + rl);
  const bf16* Arp = A + (size_t)arow * lda + ((lane >> 4) << 3);
  const bf16* We = W + (size_t)e * K * N + nbase;

  v8bf ch = *(const v8bf*)(We + (size_t)ks * N + cs);

  v8f acc0 = {}, acc1 = {}, acc2 = {}, acc3 = {};
  int buf = 0;
  for (int k0 = 0; k0 < K; k0 += 32) {
#pragma unroll
    for (int j = 0; j < 8; ++j) Bs[buf][(cs + j) * 32 + ks] = ch[j];
    __syncthreads();

    if (k0 + 32 < K)
      ch = *(const v8bf*)(We + (size_t)(k0 + 32 + ks) * N + cs);

    v16bf af;
#pragma unroll
    for (int i = 0; i < 8; ++i) af[i] = Arp[k0 + i];
#pragma unroll
    for (int i = 0; i < 8; ++i) af[8 + i] = Arp[k0 + 16 + i];

    acc0 = WMMA_BF16(af, frag_row(&Bs[buf][ 0 * 32], 32), acc0);
    acc1 = WMMA_BF16(af, frag_row(&Bs[buf][16 * 32], 32), acc1);
    acc2 = WMMA_BF16(af, frag_row(&Bs[buf][32 * 32], 32), acc2);
    acc3 = WMMA_BF16(af, frag_row(&Bs[buf][48 * 32], 32), acc3);
    buf ^= 1;
  }

  int n = lane & 15;
  const float* bp = bias + (size_t)e * N + nbase + n;
  float b0 = bp[0], b1 = bp[16], b2 = bp[32], b3 = bp[48];
#pragma unroll
  for (int r = 0; r < 8; ++r) {
    int ml = mbase + wave * 16 + r + ((lane >> 4) << 3);
    if (ml >= count) continue;
    float v0 = acc0[r] + b0, v1 = acc1[r] + b1;
    float v2 = acc2[r] + b2, v3 = acc3[r] + b3;
    if (GELU) {
      v0 = 0.5f * v0 * (1.f + erff(v0 * 0.70710678118f));
      v1 = 0.5f * v1 * (1.f + erff(v1 * 0.70710678118f));
      v2 = 0.5f * v2 * (1.f + erff(v2 * 0.70710678118f));
      v3 = 0.5f * v3 * (1.f + erff(v3 * 0.70710678118f));
    }
    int crow = SCATTER ? perm[c0 + ml] : (c0 + ml);
    size_t o = (size_t)crow * ldc + nbase + n;
    if (WF) { Cf[o] = v0; Cf[o + 16] = v1; Cf[o + 32] = v2; Cf[o + 48] = v3; }
    if (WB) { Cb[o] = (bf16)v0; Cb[o + 16] = (bf16)v1;
              Cb[o + 32] = (bf16)v2; Cb[o + 48] = (bf16)v3; }
  }
}

// ============================================================================
// Flash attention: one wave per 16-query tile.  No cross-lane shuffles:
//  - running row-max via non-returning ds_max_num_f32 atomics on a 16-entry
//    LDS array (DS ops from one wave execute in order -> no barriers needed),
//  - row-sum of P via one extra WMMA against an all-ones B fragment (the row
//    sum lands in every column with the same element<->row mapping).
// ============================================================================
__global__ void k_attn(const bf16* __restrict__ q, const bf16* __restrict__ k,
                       const bf16* __restrict__ v, bf16* __restrict__ ctx) {
  __shared__ bf16 Ps[16 * 32];   // P tile, row-major [m][k], 1 KB
  __shared__ bf16 Vs[64 * 32];   // V tile transposed [n][k], 4 KB
  __shared__ float Mx[16];       // running row max
  int qt = blockIdx.x, h = blockIdx.y, b = blockIdx.z;
  int lane = threadIdx.x & 31;
  int m0 = ((lane >> 4) << 3);
  int n = lane & 15;
  const bf16* Qp = q + (size_t)(b * SEQ + qt * 16) * DMODEL + h * DHEAD;
  const bf16* Kp = k + (size_t)b * SEQ * DMODEL + h * DHEAD;
  const bf16* Vp = v + (size_t)b * SEQ * DMODEL + h * DHEAD;
  v16bf a0 = frag_row(Qp, DMODEL);
  v16bf a1 = frag_row(Qp + 32, DMODEL);
  v16bf onesf;
#pragma unroll
  for (int i = 0; i < 16; ++i) onesf[i] = (bf16)1.0f;
  v8f o0 = {}, o1 = {}, o2 = {}, o3 = {};
  float mrun[8], lrun[8];
#pragma unroll
  for (int r = 0; r < 8; ++r) { mrun[r] = -1e30f; lrun[r] = 0.f; }
  Mx[n] = -1e30f;                // lanes 16-31 write duplicates (same value)

  for (int kt = 0; kt < SEQ; kt += 32) {
    // ---- S = Q @ K^T (K rows are contiguous -> frag_row direct) ----
    v8f s0 = {}, s1 = {};
    v16bf bk;
    bk = frag_row(Kp + (size_t)kt * DMODEL, DMODEL);             s0 = WMMA_BF16(a0, bk, s0);
    bk = frag_row(Kp + (size_t)kt * DMODEL + 32, DMODEL);        s0 = WMMA_BF16(a1, bk, s0);
    bk = frag_row(Kp + (size_t)(kt + 16) * DMODEL, DMODEL);      s1 = WMMA_BF16(a0, bk, s1);
    bk = frag_row(Kp + (size_t)(kt + 16) * DMODEL + 32, DMODEL); s1 = WMMA_BF16(a1, bk, s1);

    // ---- stage V[kt..kt+32) x 64 transposed into LDS ----
#pragma unroll
    for (int r = 0; r < 8; ++r) {
      int c = r * 32 + lane;               // 256 chunks of 8 bf16
      int kk = c >> 3;
      int coff = (c & 7) << 3;
      v8bf chv = *(const v8bf*)(Vp + (size_t)(kt + kk) * DMODEL + coff);
#pragma unroll
      for (int j = 0; j < 8; ++j) Vs[(coff + j) * 32 + kk] = chv[j];
    }

    // ---- running row max via LDS atomics (no shuffles, no waits) ----
    float x0[8], x1[8];
#pragma unroll
    for (int r = 0; r < 8; ++r) {
      x0[r] = s0[r] * 0.125f;              // 1/sqrt(64)
      x1[r] = s1[r] * 0.125f;
      atomicMax(&Mx[m0 + r], fmaxf(x0[r], x1[r]));
    }
    float corr[8];
#pragma unroll
    for (int r = 0; r < 8; ++r) {
      float mnew = Mx[m0 + r];             // running max (>= mrun)
      corr[r] = __expf(mrun[r] - mnew);
      mrun[r] = mnew;
      Ps[(m0 + r) * 32 + n]      = (bf16)__expf(x0[r] - mnew);
      Ps[(m0 + r) * 32 + 16 + n] = (bf16)__expf(x1[r] - mnew);
      o0[r] *= corr[r]; o1[r] *= corr[r]; o2[r] *= corr[r]; o3[r] *= corr[r];
    }
    __syncthreads();
    v16bf pf = frag_row((const bf16*)Ps, 32);
    v8f zero = {};
    v8f psum = WMMA_BF16(pf, onesf, zero); // row sum of P in every column
#pragma unroll
    for (int r = 0; r < 8; ++r) lrun[r] = lrun[r] * corr[r] + psum[r];
    o0 = WMMA_BF16(pf, frag_row(&Vs[ 0 * 32], 32), o0);
    o1 = WMMA_BF16(pf, frag_row(&Vs[16 * 32], 32), o1);
    o2 = WMMA_BF16(pf, frag_row(&Vs[32 * 32], 32), o2);
    o3 = WMMA_BF16(pf, frag_row(&Vs[48 * 32], 32), o3);
    __syncthreads();
  }
#pragma unroll
  for (int r = 0; r < 8; ++r) {
    float inv = 1.f / lrun[r];
    int row = b * SEQ + qt * 16 + m0 + r;
    bf16* crow = ctx + (size_t)row * DMODEL + h * DHEAD;
    crow[n]      = (bf16)(o0[r] * inv);
    crow[16 + n] = (bf16)(o1[r] * inv);
    crow[32 + n] = (bf16)(o2[r] * inv);
    crow[48 + n] = (bf16)(o3[r] * inv);
  }
}

// ---------------- residual + LayerNorm (one wave per token) -----------------
__global__ void k_add_ln(const float* __restrict__ a, const float* __restrict__ b,
                         const float* __restrict__ g, const float* __restrict__ beta,
                         float* __restrict__ outf, bf16* __restrict__ outb) {
  int t = blockIdx.x;
  int lane = threadIdx.x & 31;
  const float* ar = a + (size_t)t * DMODEL;
  const float* br = b + (size_t)t * DMODEL;
  float vals[16];
  float s = 0.f;
#pragma unroll
  for (int i = 0; i < 16; ++i) {
    float v = ar[lane + 32 * i] + br[lane + 32 * i];
    vals[i] = v; s += v;
  }
  for (int o = 1; o < 32; o <<= 1) s += __shfl_xor(s, o, 32);
  float mu = s * (1.f / DMODEL);
  float vs = 0.f;
#pragma unroll
  for (int i = 0; i < 16; ++i) { float d = vals[i] - mu; vs += d * d; }
  for (int o = 1; o < 32; o <<= 1) vs += __shfl_xor(vs, o, 32);
  float rstd = rsqrtf(vs * (1.f / DMODEL) + EPS);
#pragma unroll
  for (int i = 0; i < 16; ++i) {
    int c = lane + 32 * i;
    float y = (vals[i] - mu) * rstd * g[c] + beta[c];
    if (outf) outf[(size_t)t * DMODEL + c] = y;
    if (outb) outb[(size_t)t * DMODEL + c] = (bf16)y;
  }
}

// ---------------- gate logits + argmax (one thread per token) ---------------
__global__ void k_gate(const float* __restrict__ xln, const float* __restrict__ Wg,
                       const float* __restrict__ bg, int* __restrict__ idx) {
  int t = blockIdx.x * blockDim.x + threadIdx.x;
  if (t >= NTOK) return;
  const float* xr = xln + (size_t)t * DMODEL;
  float best = -1e30f; int bi = 0;
  for (int e = 0; e < NEXP; ++e) {
    float s = bg[e];
    for (int c = 0; c < DMODEL; ++c) s += xr[c] * Wg[c * NEXP + e];
    if (s > best) { best = s; bi = e; }   // first-max like jnp.argmax
  }
  idx[t] = bi;
}

// ---------------- routing helpers ------------------------------------------
__global__ void k_zero_i32(int* p, int n) {
  int i = blockIdx.x * blockDim.x + threadIdx.x;
  if (i < n) p[i] = 0;
}
__global__ void k_count(const int* __restrict__ idx, int* __restrict__ cnt) {
  int t = blockIdx.x * blockDim.x + threadIdx.x;
  if (t < NTOK) atomicAdd(&cnt[idx[t]], 1);
}
__global__ void k_offsets(const int* __restrict__ cnt, int* __restrict__ offs) {
  if (threadIdx.x == 0) {
    int s = 0;
    for (int e = 0; e < NEXP; ++e) { offs[e] = s; s += cnt[e]; }
  }
}
__global__ void k_scatter(const int* __restrict__ idx, const int* __restrict__ offs,
                          int* __restrict__ fill, int* __restrict__ perm) {
  int t = blockIdx.x * blockDim.x + threadIdx.x;
  if (t >= NTOK) return;
  int e = idx[t];
  int p = atomicAdd(&fill[e], 1);
  perm[offs[e] + p] = t;
}

// ---------------- host orchestration ----------------------------------------
extern "C" void kernel_launch(void* const* d_in, const int* in_sizes, int n_in,
                              void* d_out, int out_size, void* d_ws, size_t ws_size,
                              hipStream_t stream) {
  const float* x    = (const float*)d_in[0];
  const float* Wq   = (const float*)d_in[1];
  const float* bq   = (const float*)d_in[2];
  const float* Wk   = (const float*)d_in[3];
  const float* bk   = (const float*)d_in[4];
  const float* Wv   = (const float*)d_in[5];
  const float* bv   = (const float*)d_in[6];
  const float* Wo   = (const float*)d_in[7];
  const float* bo   = (const float*)d_in[8];
  const float* g1   = (const float*)d_in[9];
  const float* be1  = (const float*)d_in[10];
  const float* g2   = (const float*)d_in[11];
  const float* be2  = (const float*)d_in[12];
  const float* Wg   = (const float*)d_in[13];
  const float* bg   = (const float*)d_in[14];
  const float* W1   = (const float*)d_in[15];
  const float* b1   = (const float*)d_in[16];
  const float* W2   = (const float*)d_in[17];
  const float* b2   = (const float*)d_in[18];

  char* w = (char*)d_ws;
  auto alloc = [&](size_t bytes) -> char* {
    char* p = w;
    w += (bytes + 255) & ~(size_t)255;
    return p;
  };
  const size_t TD  = (size_t)NTOK * DMODEL;
  const size_t DD  = (size_t)DMODEL * DMODEL;
  const size_t EKH = (size_t)NEXP * DMODEL * HID;
  const size_t TH  = (size_t)NTOK * HID;

  bf16* xb   = (bf16*)alloc(TD * 2);
  bf16* wqb  = (bf16*)alloc(DD * 2);
  bf16* wkb  = (bf16*)alloc(DD * 2);
  bf16* wvb  = (bf16*)alloc(DD * 2);
  bf16* wob  = (bf16*)alloc(DD * 2);
  bf16* w1b  = (bf16*)alloc(EKH * 2);
  bf16* w2b  = (bf16*)alloc(EKH * 2);
  bf16* qb   = (bf16*)alloc(TD * 2);
  bf16* kbuf = (bf16*)alloc(TD * 2);
  bf16* vbuf = (bf16*)alloc(TD * 2);
  bf16* ctxb = (bf16*)alloc(TD * 2);
  float* saf = (float*)alloc(TD * 4);
  float* ln1f= (float*)alloc(TD * 4);
  bf16* ln1b = (bf16*)alloc(TD * 2);
  int*  idx  = (int*)alloc(NTOK * 4);
  int*  cnt  = (int*)alloc(NEXP * 4);
  int*  offs = (int*)alloc(NEXP * 4);
  int*  fill = (int*)alloc(NEXP * 4);
  int*  perm = (int*)alloc(NTOK * 4);
  bf16* hb   = (bf16*)alloc(TH * 2);
  float* moef= (float*)alloc(TD * 4);

  auto cvt = [&](const float* src, bf16* dst, size_t n) {
    int n8 = (int)(n / 8);
    k_cvt8<<<dim3((n8 + 255) / 256), 256, 0, stream>>>(src, dst, n8);
  };
  cvt(x,  xb,  TD);
  cvt(Wq, wqb, DD);  cvt(Wk, wkb, DD);  cvt(Wv, wvb, DD);  cvt(Wo, wob, DD);
  cvt(W1, w1b, EKH); cvt(W2, w2b, EKH);

  // Q,K,V projections (bf16 out): 128x64 tiles, 256 threads/block
  dim3 gproj(DMODEL / 64, NTOK / 128);
  k_gemm_t<false, true><<<gproj, 256, 0, stream>>>(xb, wqb, bq, nullptr, qb,   DMODEL, DMODEL);
  k_gemm_t<false, true><<<gproj, 256, 0, stream>>>(xb, wkb, bk, nullptr, kbuf, DMODEL, DMODEL);
  k_gemm_t<false, true><<<gproj, 256, 0, stream>>>(xb, wvb, bv, nullptr, vbuf, DMODEL, DMODEL);

  // flash attention
  k_attn<<<dim3(SEQ / 16, NHEAD, BATCH), 32, 0, stream>>>(qb, kbuf, vbuf, ctxb);

  // output projection (fp32 out)
  k_gemm_t<true, false><<<gproj, 256, 0, stream>>>(ctxb, wob, bo, saf, nullptr, DMODEL, DMODEL);

  // x = LN(x + sa)
  k_add_ln<<<NTOK, 32, 0, stream>>>(x, saf, g1, be1, ln1f, ln1b);

  // gate + routing
  k_gate<<<NTOK / 256, 256, 0, stream>>>(ln1f, Wg, bg, idx);
  k_zero_i32<<<1, 32, 0, stream>>>(cnt, NEXP);
  k_zero_i32<<<1, 32, 0, stream>>>(fill, NEXP);
  k_count<<<NTOK / 256, 256, 0, stream>>>(idx, cnt);
  k_offsets<<<1, 1, 0, stream>>>(cnt, offs);
  k_scatter<<<NTOK / 256, 256, 0, stream>>>(idx, offs, fill, perm);

  // grouped MoE: h = gelu(x[perm] @ W1[e] + b1[e]) ; moe[perm] = h @ W2[e] + b2[e]
  k_moe_gemm_t<false, true, true, true, false>
      <<<dim3(HID / 64, NTOK / 128, NEXP), 256, 0, stream>>>(
      ln1b, DMODEL, w1b, DMODEL, HID, b1, nullptr, hb, HID, perm, offs, cnt);
  k_moe_gemm_t<true, false, false, false, true>
      <<<dim3(DMODEL / 64, NTOK / 128, NEXP), 256, 0, stream>>>(
      hb, HID, w2b, HID, DMODEL, b2, moef, nullptr, DMODEL, perm, offs, cnt);

  // out = LN(x + moe)
  k_add_ln<<<NTOK, 32, 0, stream>>>(ln1f, moef, g2, be2, (float*)d_out, nullptr);
}
